// Cffn_29721173688896
// MI455X (gfx1250) — matrix operand order repaired and checked by reading, not compiled, not run on hardware
//
#include <hip/hip_runtime.h>

// ---------------------------------------------------------------------------
// Fused: linear_out = x @ U_w^T ; g = sigmoid(x @ gate_w^T) * x ;
//        z = continued_fraction(g * ladder_w) ; out = linear + sum_l z_l*V[:,l]
// GEMMs via v_wmma_f32_16x16x32_bf16 with bf16 hi/lo split (3-product scheme).
// Register-prefetch software pipeline on the K loop; ladder/V tables fetched
// once per block via the Tensor Data Mover (tensor_load_to_lds, TENSORcnt).
// ---------------------------------------------------------------------------

typedef __bf16 bf16;
typedef __attribute__((ext_vector_type(16))) __bf16 v16bf;
typedef __attribute__((ext_vector_type(8)))  __bf16 v8bf;
typedef __attribute__((ext_vector_type(8)))  float  v8f;
typedef __attribute__((ext_vector_type(4)))  unsigned int u32x4;
typedef __attribute__((ext_vector_type(8)))  int         i32x8;
typedef __attribute__((ext_vector_type(4)))  int         i32x4;

#define DIMN      2048
#define M_TOT     4096
#define NLAD      3
#define DEPTH     5
#define EPS_CF    0.01f

#define BM        128
#define BN        128
#define BK        32
#define NK        (DIMN / BK)
#define LDK       40          // padded LDS row length (bf16 elems): 80B, 16B-aligned
#define THREADS   512         // 16 waves of 32

#if defined(__has_builtin)
#if __has_builtin(__builtin_amdgcn_tensor_load_to_lds)
#define HAVE_TDM 1
#endif
#endif

#define WMMA_BF16(A, B, C) \
    __builtin_amdgcn_wmma_f32_16x16x32_bf16(false, (A), false, (B), (short)0, (C), false, false)

#ifdef HAVE_TDM
// Issue one TDM 2-D tile load (fp32 elements): rows x cols from global into
// contiguous LDS.  D# layout per CDNA5 ISA.
__device__ __forceinline__ void tdm_load_2d(unsigned lds_off, const void* gptr,
                                            unsigned cols, unsigned rows,
                                            unsigned long long row_stride_elems)
{
    const unsigned long long ga = (unsigned long long)(size_t)gptr;
    u32x4 g0;
    g0[0] = 1u;                                             // count=1, user desc
    g0[1] = lds_off;                                        // lds_addr (bytes)
    g0[2] = (unsigned)(ga & 0xffffffffu);                   // global_addr[31:0]
    g0[3] = (unsigned)((ga >> 32) & 0x01ffffffu) | (2u << 30); // addr[56:32] | type=2
    i32x8 g1;
    g1[0] = (int)(2u << 16);                                // wg_mask=0, data_size=4B
    g1[1] = (int)((cols & 0xffffu) << 16);                  // tensor_dim0[15:0]
    g1[2] = (int)(((cols >> 16) & 0xffffu) | ((rows & 0xffffu) << 16)); // td0[31:16] | td1[15:0]
    g1[3] = (int)(((rows >> 16) & 0xffffu) | ((cols & 0xffffu) << 16)); // td1[31:16] | tile_dim0
    g1[4] = (int)(rows & 0xffffu);                          // tile_dim1 (tile_dim2=0)
    g1[5] = (int)(unsigned)(row_stride_elems & 0xffffffffull);          // td0_stride[31:0]
    g1[6] = (int)(unsigned)((row_stride_elems >> 32) & 0xffffull);      // td0_stride[47:32]
    g1[7] = 0;
    const i32x4 z4 = {0, 0, 0, 0};
#if __has_include(<hip/amd_detail/amd_gfx1250_TDM.h>)
    const i32x8 z8 = {0, 0, 0, 0, 0, 0, 0, 0};
    __builtin_amdgcn_tensor_load_to_lds(g0, g1, z4, z4, z8, 0);
#else
    __builtin_amdgcn_tensor_load_to_lds(g0, g1, z4, z4, 0);
#endif
}
#endif

__global__ __launch_bounds__(THREADS)
void cffn_fused_kernel(const float* __restrict__ x,
                       const float* __restrict__ Uw,
                       const float* __restrict__ Gw,
                       const float* __restrict__ Lw,   // (NLAD, DIMN, DEPTH)
                       const float* __restrict__ Vw,   // (DIMN, NLAD)
                       float* __restrict__ out)
{
    __shared__ __align__(16) bf16 Ah[BM * LDK];
    __shared__ __align__(16) bf16 Al[BM * LDK];
    __shared__ __align__(16) bf16 Uh[BN * LDK];
    __shared__ __align__(16) bf16 Ul[BN * LDK];
    __shared__ __align__(16) bf16 Gh[BN * LDK];
    __shared__ __align__(16) bf16 Gl[BN * LDK];
    __shared__ __align__(16) float LwS[NLAD * BN * DEPTH];  // [l][col][k]
    __shared__ __align__(16) float VwS[BN * NLAD];          // [col][l]

    const int tid  = threadIdx.x;
    const int lane = tid & 31;
    const int wave = tid >> 5;
    const int wm   = (wave >> 2) * 32;       // wave tile origin in block tile
    const int wn   = (wave & 3)  * 32;
    const int m0   = blockIdx.y * BM;        // global tile origin
    const int n0   = blockIdx.x * BN;

#ifdef HAVE_TDM
    // Kick off the per-block ladder/V table DMAs; they overlap the whole GEMM.
    if (wave == 0) {
        tdm_load_2d((unsigned)(size_t)(void*)LwS, Lw + (size_t)n0 * DEPTH,
                    BN * DEPTH, NLAD, (unsigned long long)DIMN * DEPTH);
        tdm_load_2d((unsigned)(size_t)(void*)VwS, Vw + (size_t)n0 * NLAD,
                    BN * NLAD, 1, (unsigned long long)BN * NLAD);
    }
#endif

    v8f cU[2][2], cG[2][2];
#pragma unroll
    for (int i = 0; i < 2; ++i)
#pragma unroll
        for (int j = 0; j < 2; ++j) { cU[i][j] = {}; cG[i][j] = {}; }

    // ---- register-prefetch staging pipeline -------------------------------
    float4 rx[2], ru[2], rg[2];
    auto gload = [&](int k0) {
#pragma unroll
        for (int it = 0; it < 2; ++it) {
            const int idx = tid + it * THREADS;     // float4 index 0..1023
            const int r   = idx >> 3;               // 0..127
            const int c4  = (idx & 7) << 2;         // 0,4,..,28
            rx[it] = *(const float4*)(x  + (size_t)(m0 + r) * DIMN + k0 + c4);
            ru[it] = *(const float4*)(Uw + (size_t)(n0 + r) * DIMN + k0 + c4);
            rg[it] = *(const float4*)(Gw + (size_t)(n0 + r) * DIMN + k0 + c4);
        }
    };
    auto conv_store = [&](const float4* regs, bf16* dh, bf16* dl) {
#pragma unroll
        for (int it = 0; it < 2; ++it) {
            const int idx = tid + it * THREADS;
            const int r   = idx >> 3;
            const int c4  = (idx & 7) << 2;
            const float v[4] = { regs[it].x, regs[it].y, regs[it].z, regs[it].w };
#pragma unroll
            for (int e = 0; e < 4; ++e) {
                const bf16 h = (bf16)v[e];
                const bf16 l = (bf16)(v[e] - (float)h);
                dh[r * LDK + c4 + e] = h;
                dl[r * LDK + c4 + e] = l;
            }
        }
    };

    // A fragment (16x32 bf16): lane holds M=lane%16; K chunks {h*8..h*8+7, 16+h*8..}
    auto loadA = [&](const bf16* s, int moff) -> v16bf {
        const int row = moff + (lane & 15);
        const int kb  = (lane >> 4) * 8;
        const v8bf c0 = *(const v8bf*)(s + row * LDK + kb);
        const v8bf c1 = *(const v8bf*)(s + row * LDK + kb + 16);
        v16bf r;
#pragma unroll
        for (int i = 0; i < 8; ++i) { r[i] = c0[i]; r[i + 8] = c1[i]; }
        return r;
    };
    // B fragment (32x16 bf16): lane holds N=lane%16; K chunks {h*8..h*8+7, 16+h*8..}
    auto loadB = [&](const bf16* s, int noff) -> v16bf {
        const int row = noff + (lane & 15);
        const int kb  = (lane >> 4) * 8;
        const v8bf c0 = *(const v8bf*)(s + row * LDK + kb);
        const v8bf c1 = *(const v8bf*)(s + row * LDK + kb + 16);
        v16bf r;
#pragma unroll
        for (int i = 0; i < 8; ++i) { r[i] = c0[i]; r[i + 8] = c1[i]; }
        return r;
    };

    gload(0);                                     // prime the pipeline

    for (int kt = 0; kt < NK; ++kt) {
        __syncthreads();                          // prior LDS reads complete
        conv_store(rx, Ah, Al);
        conv_store(ru, Uh, Ul);
        conv_store(rg, Gh, Gl);
        if (kt + 1 < NK) gload((kt + 1) * BK);    // next tile in flight during compute
        __syncthreads();                          // LDS tiles visible

        v16bf aH[2], aL[2];
#pragma unroll
        for (int mi = 0; mi < 2; ++mi) {
            aH[mi] = loadA(Ah, wm + 16 * mi);
            aL[mi] = loadA(Al, wm + 16 * mi);
        }
#pragma unroll
        for (int ni = 0; ni < 2; ++ni) {
            const v16bf bUh = loadB(Uh, wn + 16 * ni);
            const v16bf bUl = loadB(Ul, wn + 16 * ni);
            const v16bf bGh = loadB(Gh, wn + 16 * ni);
            const v16bf bGl = loadB(Gl, wn + 16 * ni);
#pragma unroll
            for (int mi = 0; mi < 2; ++mi) {
                // hi*hi + hi*lo + lo*hi  (fp32 accumulate)
                cU[mi][ni] = WMMA_BF16(aH[mi], bUh, cU[mi][ni]);
                cU[mi][ni] = WMMA_BF16(aH[mi], bUl, cU[mi][ni]);
                cU[mi][ni] = WMMA_BF16(aL[mi], bUh, cU[mi][ni]);
                cG[mi][ni] = WMMA_BF16(aH[mi], bGh, cG[mi][ni]);
                cG[mi][ni] = WMMA_BF16(aH[mi], bGl, cG[mi][ni]);
                cG[mi][ni] = WMMA_BF16(aL[mi], bGh, cG[mi][ni]);
            }
        }
    }

#ifdef HAVE_TDM
    if (wave == 0) __builtin_amdgcn_s_wait_tensorcnt(0);   // table DMAs done
#endif
    __syncthreads();                                        // tables visible to all

    // ---- fused epilogue: gate, continued fraction, ladder sum, add, store ----
    const int nLane  = lane & 15;
    const int rowSel = (lane >> 4) * 8;     // C layout: lanes>=16 hold M rows +8

#pragma unroll
    for (int ni = 0; ni < 2; ++ni) {
        const int colL = wn + 16 * ni + nLane;         // column within block tile
        const int gn   = n0 + colL;                    // global output column
        float lw[NLAD][DEPTH], vv[NLAD];
#pragma unroll
        for (int l = 0; l < NLAD; ++l) {
#ifdef HAVE_TDM
            vv[l] = VwS[colL * NLAD + l];
#pragma unroll
            for (int k = 0; k < DEPTH; ++k)
                lw[l][k] = LwS[(l * BN + colL) * DEPTH + k];
#else
            vv[l] = Vw[gn * NLAD + l];
#pragma unroll
            for (int k = 0; k < DEPTH; ++k)
                lw[l][k] = Lw[((size_t)l * DIMN + gn) * DEPTH + k];
#endif
        }
#pragma unroll
        for (int mi = 0; mi < 2; ++mi) {
#pragma unroll
            for (int j = 0; j < 8; ++j) {
                const int gm  = m0 + wm + 16 * mi + rowSel + j;   // output row
                const float lin  = cU[mi][ni][j];
                const float gacc = cG[mi][ni][j];
                const float xv   = x[(size_t)gm * DIMN + gn];
                const float g    = xv / (1.0f + __expf(-gacc));   // sigmoid(gacc)*x

                float comb = 0.0f;
#pragma unroll
                for (int l = 0; l < NLAD; ++l) {
                    float z = g * lw[l][DEPTH - 1];
#pragma unroll
                    for (int k = DEPTH - 2; k >= 0; --k) {
                        float den = 1.0f + z;
                        if (fabsf(den) < EPS_CF) den = (den >= 0.0f) ? EPS_CF : -EPS_CF;
                        z = (g * lw[l][k]) / den;
                    }
                    comb += z * vv[l];
                }
                out[(size_t)gm * DIMN + gn] = lin + comb;
            }
        }
    }
}

extern "C" void kernel_launch(void* const* d_in, const int* in_sizes, int n_in,
                              void* d_out, int out_size, void* d_ws, size_t ws_size,
                              hipStream_t stream)
{
    (void)in_sizes; (void)n_in; (void)out_size; (void)d_ws; (void)ws_size;
    const float* x  = (const float*)d_in[0];
    const float* Uw = (const float*)d_in[1];
    const float* Gw = (const float*)d_in[2];
    const float* Lw = (const float*)d_in[3];
    const float* Vw = (const float*)d_in[4];
    float* out = (float*)d_out;

    dim3 grid(DIMN / BN, M_TOT / BM);   // (16, 32)
    cffn_fused_kernel<<<grid, THREADS, 0, stream>>>(x, Uw, Gw, Lw, Vw, out);
}
